// RobertaAttention_40948218200349
// MI455X (gfx1250) — matrix operand order repaired and checked by reading, not compiled, not run on hardware
//
#include <hip/hip_runtime.h>
#include <hip/hip_bf16.h>

typedef __attribute__((ext_vector_type(16))) _Float16 v16h;
typedef __attribute__((ext_vector_type(8)))  _Float16 v8h;
typedef __attribute__((ext_vector_type(8)))  float    v8f;
typedef __attribute__((ext_vector_type(4))) unsigned int u32x4;
typedef __attribute__((ext_vector_type(8))) int          i32x8;
typedef __attribute__((ext_vector_type(4))) int          i32x4;

#define BATCH  4
#define SEQ    2048
#define DMODEL 768
#define NH     12
#define DH     64
#define NROWS  (BATCH * SEQ)          // 8192

union FragU { v16h v; v8h h[2]; };

// Issue a 2D TDM load of an f16 tile [rows x cols] (cols contiguous, row
// stride `srcStride` halfs) into LDS at byte offset `ldsAddr`, with LDS row
// padding of 4 DWORDs every 2^(padIntvl+1) DWORDs.  D# per ISA 08 §8.
// This toolchain exposes the 6-arg builtin (extra int32x8 beyond the four
// architectural D# groups); pass it zero-filled.
__device__ __forceinline__ void tdm_load_f16_tile(
    const _Float16* gsrc, unsigned ldsAddr,
    unsigned tensorDim0, unsigned tensorDim1, unsigned srcStride,
    unsigned tileDim0, unsigned tileDim1, unsigned padIntvl /*log2(dw)-1*/)
{
    const unsigned long long ga = (unsigned long long)(size_t)gsrc;
    u32x4 g0;
    g0[0] = 1u;                                   // count=1, user D#, no gather
    g0[1] = ldsAddr;                              // lds_addr (bytes)
    g0[2] = (unsigned)(ga & 0xffffffffu);         // global_addr[31:0]
    g0[3] = (unsigned)((ga >> 32) & 0x01ffffffu)  // global_addr[56:32]
          | (2u << 30);                           // type = 2 ("image")
    i32x8 g1;
    g1[0] = (int)((1u << 16)                      // data_size = 2 bytes
          |       (1u << 20)                      // pad_enable
          |       (padIntvl << 22)                // pad_interval
          |       (3u << 25));                    // pad_amount = 4 DWORDs
    g1[1] = (int)(tensorDim0 << 16);              // tensor_dim0 (bits 79:48)
    g1[2] = (int)(tensorDim1 << 16);              // tensor_dim1 (bits 111:80)
    g1[3] = (int)(tileDim0 << 16);                // tile_dim0  (bits 127:112)
    g1[4] = (int)tileDim1;                        // tile_dim1; tile_dim2=0
    g1[5] = (int)srcStride;                       // tensor_dim0_stride lo
    g1[6] = 0;                                    // stride hi | dim1_stride lo
    g1[7] = 0;
    i32x4 g2; g2[0] = 1; g2[1] = 1; g2[2] = 0; g2[3] = 0;   // dim2=dim3=1 (benign)
    i32x4 g3; g3[0] = 0; g3[1] = 1 << 16; g3[2] = 0; g3[3] = 0; // tensor_dim4=1
    i32x8 g4 = {0, 0, 0, 0, 0, 0, 0, 0};          // non-architectural extra arg
    __builtin_amdgcn_tensor_load_to_lds(g0, g1, g2, g3, g4, 0);
}

// ---------------------------------------------------------------------------
// Kernel 1: fused QKV projection.  q = X @ W^T + b, split to [B,H,S,64] f16.
// Block: 256 threads (8 waves). Tile: M=128, N=64 (one head), Kstep=32.
// grid = (12 heads, 64 row-tiles, 3 weights).  f32 sources need cvt -> manual
// staging (TDM cannot convert), with constant-trip unrolled loops.
// ---------------------------------------------------------------------------
__global__ __launch_bounds__(256) void rba_qkv_kernel(
    const float* __restrict__ X,
    const float* __restrict__ Wq, const float* __restrict__ bq,
    const float* __restrict__ Wk, const float* __restrict__ bk,
    const float* __restrict__ Wv, const float* __restrict__ bv,
    _Float16* __restrict__ qo, _Float16* __restrict__ ko, _Float16* __restrict__ vo)
{
    const float* W; const float* bias; _Float16* out;
    if (blockIdx.z == 0)      { W = Wq; bias = bq; out = qo; }
    else if (blockIdx.z == 1) { W = Wk; bias = bk; out = ko; }
    else                      { W = Wv; bias = bv; out = vo; }

    const int h  = blockIdx.x;          // head index == N tile of 64
    const int m0 = blockIdx.y * 128;

    __shared__ _Float16 Xs[128 * 40];   // 128 rows x 32 K (stride 40, 16B-aligned)
    __shared__ _Float16 Ws[64 * 40];    // 64 cols(n) x 32 K

    const int tid  = threadIdx.x;
    const int wave = tid >> 5;
    const int lane = tid & 31;
    const int hi   = lane >> 4;
    const int l15  = lane & 15;
    const int kb   = hi * 8;            // A-fragment K base (interleaved halves)
    const int khalf = hi * 16;          // contiguous-half K base (unused; B also uses kb)

    v8f acc[4] = {};

    for (int k0 = 0; k0 < DMODEL; k0 += 32) {
        __syncthreads();
        // Stage X tile 128x32 f32 -> f16: 1024 float4 groups, 4 per thread.
        #pragma unroll
        for (int it = 0; it < 4; ++it) {
            const int i = it * 256 + tid;
            const int r = i >> 3, c4 = (i & 7) << 2;
            const float4 f = *(const float4*)&X[(size_t)(m0 + r) * DMODEL + k0 + c4];
            _Float16* dst = &Xs[r * 40 + c4];
            dst[0] = (_Float16)f.x; dst[1] = (_Float16)f.y;
            dst[2] = (_Float16)f.z; dst[3] = (_Float16)f.w;
        }
        // Stage W tile 64x32 f32 -> f16: 512 groups, 2 per thread.
        #pragma unroll
        for (int it = 0; it < 2; ++it) {
            const int i = it * 256 + tid;
            const int r = i >> 3, c4 = (i & 7) << 2;
            const float4 f = *(const float4*)&W[(size_t)(h * DH + r) * DMODEL + k0 + c4];
            _Float16* dst = &Ws[r * 40 + c4];
            dst[0] = (_Float16)f.x; dst[1] = (_Float16)f.y;
            dst[2] = (_Float16)f.z; dst[3] = (_Float16)f.w;
        }
        // prefetch next K-step tiles into cache (global_prefetch_b8)
        if (k0 + 32 < DMODEL) {
            __builtin_prefetch(&X[(size_t)(m0 + (tid >> 1)) * DMODEL + k0 + 32 + (tid & 1) * 16], 0, 1);
            __builtin_prefetch(&W[(size_t)(h * DH + (tid & 63)) * DMODEL + k0 + 32], 0, 1);
        }
        __syncthreads();

        // A fragment: row = 16*wave + l15, K interleaved [kb..kb+7 | 16+kb..]
        FragU a;
        const int arow = wave * 16 + l15;
        a.h[0] = *(const v8h*)&Xs[arow * 40 + kb];
        a.h[1] = *(const v8h*)&Xs[arow * 40 + 16 + kb];

        #pragma unroll
        for (int t = 0; t < 4; ++t) {
            FragU b;
            const int bn = t * 16 + l15;
            b.h[0] = *(const v8h*)&Ws[bn * 40 + kb];
            b.h[1] = *(const v8h*)&Ws[bn * 40 + 16 + kb];
            acc[t] = __builtin_amdgcn_wmma_f32_16x16x32_f16(
                false, a.v, false, b.v, (short)0, acc[t], false, false);
        }
    }

    // Epilogue: C layout row = j + 8*hi, col = 16t + l15; store head-major f16.
    #pragma unroll
    for (int t = 0; t < 4; ++t) {
        #pragma unroll
        for (int j = 0; j < 8; ++j) {
            const int m = m0 + wave * 16 + j + hi * 8;
            const int d = t * 16 + l15;
            const int b = m >> 11, s = m & (SEQ - 1);
            const float val = acc[t][j] + bias[h * DH + d];
            out[(((size_t)(b * NH + h) * SEQ) + s) * DH + d] = (_Float16)val;
        }
    }
}

// ---------------------------------------------------------------------------
// Kernel 2: flash attention. Block = 128 threads (4 waves), each wave owns a
// 16-row q tile; block covers 64 q rows. Keys in chunks of 32.  K chunk is
// staged by the Tensor Data Mover (wave 0 issues, TENSORcnt-tracked) with
// hardware LDS padding giving the stride-72 layout; V transpose stays manual
// and overlaps the DMA.   grid = (S/64 = 32, B*H = 48)
// ---------------------------------------------------------------------------
__global__ __launch_bounds__(128) void rba_attn_kernel(
    const _Float16* __restrict__ qm, const _Float16* __restrict__ km,
    const _Float16* __restrict__ vm, const float* __restrict__ mask,
    _Float16* __restrict__ ctx)
{
    const int bh = blockIdx.y;
    const int b  = bh / NH;
    const int h  = bh % NH;
    const int q0 = blockIdx.x * 64;

    const int tid  = threadIdx.x;
    const int wave = tid >> 5;
    const int lane = tid & 31;
    const int hi   = lane >> 4;
    const int l15  = lane & 15;
    const int kb   = hi * 8;
    const int khalf = hi * 16;

    __shared__ _Float16 Ks[32 * 72];       // [key][d], stride 72 (TDM-padded)
    __shared__ _Float16 Vt[64 * 40];       // transposed: [d][key], stride 40
    __shared__ _Float16 Ps[4][16 * 40];    // per-wave P tile [row][key], stride 40

    const _Float16* qh = qm + (size_t)bh * SEQ * DH;
    const _Float16* kh = km + (size_t)bh * SEQ * DH;
    const _Float16* vh = vm + (size_t)bh * SEQ * DH;
    const unsigned ksLds = (unsigned)(size_t)(void*)&Ks[0];  // LDS byte offset

    // Q fragments (K-dim = 64 -> two 16x32 A fragments), loaded once.
    const int qrow = q0 + wave * 16 + l15;
    FragU qa[2];
    #pragma unroll
    for (int f = 0; f < 2; ++f) {
        qa[f].h[0] = *(const v8h*)&qh[(size_t)qrow * DH + f * 32 + kb];
        qa[f].h[1] = *(const v8h*)&qh[(size_t)qrow * DH + f * 32 + 16 + kb];
    }

    float mrun[8], lrun[8];
    #pragma unroll
    for (int j = 0; j < 8; ++j) { mrun[j] = -1e30f; lrun[j] = 0.0f; }
    v8f acc[4] = {};

    for (int c0 = 0; c0 < SEQ; c0 += 32) {
        __syncthreads();   // previous chunk's consumers done before overwrite
        // TDM: K chunk [32 keys][64 d], row = 32 DWORDs -> pad_interval=4,
        // +4 DWORDs pad -> LDS stride 72 halfs, matching fragment reads.
        if (wave == 0)
            tdm_load_f16_tile(kh + (size_t)c0 * DH, ksLds,
                              /*tdim0=*/DH, /*tdim1=*/SEQ, /*stride=*/DH,
                              /*tile0=*/DH, /*tile1=*/32, /*padIntvl=*/4);
        // Meanwhile all waves stage V chunk transposed: v[key][d] -> Vt[d][key]
        #pragma unroll
        for (int it = 0; it < 2; ++it) {
            const int i = it * 128 + tid;
            const int r = i >> 3, c8 = (i & 7) << 3;
            v8h vv = *(const v8h*)&vh[(size_t)(c0 + r) * DH + c8];
            #pragma unroll
            for (int e = 0; e < 8; ++e) Vt[(c8 + e) * 40 + r] = vv[e];
        }
        if (wave == 0) __builtin_amdgcn_s_wait_tensorcnt(0);
        __syncthreads();

        // Scores: S16x32 = Q(16x64) @ K^T(64x32) -> 2 tiles x 2 K-steps = 4 WMMA
        v8f sc[2] = {};
        #pragma unroll
        for (int t = 0; t < 2; ++t) {
            const int n = t * 16 + l15;
            #pragma unroll
            for (int f = 0; f < 2; ++f) {
                FragU bk_;
                bk_.h[0] = *(const v8h*)&Ks[n * 72 + f * 32 + kb];
                bk_.h[1] = *(const v8h*)&Ks[n * 72 + f * 32 + 16 + kb];
                sc[t] = __builtin_amdgcn_wmma_f32_16x16x32_f16(
                    false, qa[f].v, false, bk_.v, (short)0, sc[t], false, false);
            }
        }
        const float mk0 = mask[(size_t)b * SEQ + c0 + l15];
        const float mk1 = mask[(size_t)b * SEQ + c0 + 16 + l15];

        // Online softmax per row j (row = j + 8*hi; all 16 lanes of a half
        // carry that row, so xor-reductions over masks 1..8 are row-local).
        #pragma unroll
        for (int j = 0; j < 8; ++j) {
            float x0 = sc[0][j] * 0.125f + mk0;
            float x1 = sc[1][j] * 0.125f + mk1;
            float mx = fmaxf(x0, x1);
            #pragma unroll
            for (int off = 1; off <= 8; off <<= 1) mx = fmaxf(mx, __shfl_xor(mx, off, 32));
            const float mnew  = fmaxf(mrun[j], mx);
            const float alpha = __expf(mrun[j] - mnew);
            const float p0 = __expf(x0 - mnew);
            const float p1 = __expf(x1 - mnew);
            float rs = p0 + p1;
            #pragma unroll
            for (int off = 1; off <= 8; off <<= 1) rs += __shfl_xor(rs, off, 32);
            lrun[j] = lrun[j] * alpha + rs;
            mrun[j] = mnew;
            #pragma unroll
            for (int t = 0; t < 4; ++t) acc[t][j] *= alpha;
            const int prow = j + hi * 8;
            Ps[wave][prow * 40 + l15]      = (_Float16)p0;
            Ps[wave][prow * 40 + 16 + l15] = (_Float16)p1;
        }
        // Wave-private LDS roundtrip: ensure DS stores landed before re-read.
        asm volatile("s_wait_dscnt 0x0" ::: "memory");

        // ctx(16x64) += P(16x32) @ V(32x64): 4 WMMA
        FragU pa;
        pa.h[0] = *(const v8h*)&Ps[wave][l15 * 40 + kb];
        pa.h[1] = *(const v8h*)&Ps[wave][l15 * 40 + 16 + kb];
        #pragma unroll
        for (int t = 0; t < 4; ++t) {
            FragU bv_;
            const int n = t * 16 + l15;
            bv_.h[0] = *(const v8h*)&Vt[n * 40 + kb];
            bv_.h[1] = *(const v8h*)&Vt[n * 40 + 16 + kb];
            acc[t] = __builtin_amdgcn_wmma_f32_16x16x32_f16(
                false, pa.v, false, bv_.v, (short)0, acc[t], false, false);
        }
    }

    // Epilogue: normalize by l, store ctx as f16 in [B*S, 768] for the proj GEMM.
    #pragma unroll
    for (int t = 0; t < 4; ++t) {
        #pragma unroll
        for (int j = 0; j < 8; ++j) {
            const int qr = q0 + wave * 16 + j + hi * 8;
            const int d  = t * 16 + l15;
            const float val = acc[t][j] / lrun[j];
            ctx[((size_t)b * SEQ + qr) * DMODEL + h * DH + d] = (_Float16)val;
        }
    }
}

// ---------------------------------------------------------------------------
// Kernel 3: output projection + bias + residual -> f32 pre-LN activation.
// A-tile (ctx, already f16) staged by TDM with hardware padding -> stride 40;
// W tile (f32) staged manually.  grid = (12, 64), 256 threads.
// ---------------------------------------------------------------------------
__global__ __launch_bounds__(256) void rba_proj_kernel(
    const _Float16* __restrict__ Cm, const float* __restrict__ Wo,
    const float* __restrict__ bo, const float* __restrict__ X,
    float* __restrict__ Y)
{
    const int n0 = blockIdx.x * 64;
    const int m0 = blockIdx.y * 128;

    __shared__ _Float16 As[128 * 40];   // stride 40 halfs (TDM-padded)
    __shared__ _Float16 Ws[64 * 40];

    const int tid  = threadIdx.x;
    const int wave = tid >> 5;
    const int lane = tid & 31;
    const int hi   = lane >> 4;
    const int l15  = lane & 15;
    const int kb   = hi * 8;
    const int khalf = hi * 16;
    const unsigned asLds = (unsigned)(size_t)(void*)&As[0];

    v8f acc[4] = {};

    for (int k0 = 0; k0 < DMODEL; k0 += 32) {
        __syncthreads();
        // TDM: A tile [128 rows][32 k] f16; row = 16 DWORDs -> pad_interval=3,
        // +4 DWORDs pad -> LDS stride 40 halfs.
        if (wave == 0)
            tdm_load_f16_tile(Cm + (size_t)m0 * DMODEL + k0, asLds,
                              /*tdim0=*/DMODEL, /*tdim1=*/NROWS, /*stride=*/DMODEL,
                              /*tile0=*/32, /*tile1=*/128, /*padIntvl=*/3);
        // W tile 64x32 f32 -> f16 (needs convert; manual), 2 groups/thread.
        #pragma unroll
        for (int it = 0; it < 2; ++it) {
            const int i = it * 256 + tid;
            const int r = i >> 3, c4 = (i & 7) << 2;
            const float4 f = *(const float4*)&Wo[(size_t)(n0 + r) * DMODEL + k0 + c4];
            _Float16* dst = &Ws[r * 40 + c4];
            dst[0] = (_Float16)f.x; dst[1] = (_Float16)f.y;
            dst[2] = (_Float16)f.z; dst[3] = (_Float16)f.w;
        }
        if (k0 + 32 < DMODEL)
            __builtin_prefetch(&Wo[(size_t)(n0 + (tid & 63)) * DMODEL + k0 + 32], 0, 1);
        if (wave == 0) __builtin_amdgcn_s_wait_tensorcnt(0);
        __syncthreads();

        FragU a;
        const int arow = wave * 16 + l15;
        a.h[0] = *(const v8h*)&As[arow * 40 + kb];
        a.h[1] = *(const v8h*)&As[arow * 40 + 16 + kb];
        #pragma unroll
        for (int t = 0; t < 4; ++t) {
            FragU b;
            const int bn = t * 16 + l15;
            b.h[0] = *(const v8h*)&Ws[bn * 40 + kb];
            b.h[1] = *(const v8h*)&Ws[bn * 40 + 16 + kb];
            acc[t] = __builtin_amdgcn_wmma_f32_16x16x32_f16(
                false, a.v, false, b.v, (short)0, acc[t], false, false);
        }
    }

    #pragma unroll
    for (int t = 0; t < 4; ++t) {
        #pragma unroll
        for (int j = 0; j < 8; ++j) {
            const int m = m0 + wave * 16 + j + hi * 8;
            const int n = n0 + t * 16 + l15;
            Y[(size_t)m * DMODEL + n] = acc[t][j] + bo[n] + X[(size_t)m * DMODEL + n];
        }
    }
}

// ---------------------------------------------------------------------------
// Kernel 4: in-place LayerNorm over last dim (768). One block per row.
// ---------------------------------------------------------------------------
__global__ __launch_bounds__(256) void rba_ln_kernel(
    float* __restrict__ Y, const float* __restrict__ gamma, const float* __restrict__ beta)
{
    const int tid = threadIdx.x;
    float* y = Y + (size_t)blockIdx.x * DMODEL;
    const float v0 = y[tid], v1 = y[tid + 256], v2 = y[tid + 512];

    __shared__ float red[8];
    float s = v0 + v1 + v2;
    #pragma unroll
    for (int off = 1; off < 32; off <<= 1) s += __shfl_xor(s, off, 32);
    if ((tid & 31) == 0) red[tid >> 5] = s;
    __syncthreads();
    float tot = 0.0f;
    #pragma unroll
    for (int i = 0; i < 8; ++i) tot += red[i];
    const float mu = tot * (1.0f / DMODEL);

    const float d0 = v0 - mu, d1 = v1 - mu, d2 = v2 - mu;
    float q = d0 * d0 + d1 * d1 + d2 * d2;
    #pragma unroll
    for (int off = 1; off < 32; off <<= 1) q += __shfl_xor(q, off, 32);
    __syncthreads();                       // all reads of red[] done
    if ((tid & 31) == 0) red[tid >> 5] = q;
    __syncthreads();
    float vtot = 0.0f;
    #pragma unroll
    for (int i = 0; i < 8; ++i) vtot += red[i];
    const float rstd = rsqrtf(vtot * (1.0f / DMODEL) + 1e-12f);

    y[tid]       = d0 * rstd * gamma[tid]       + beta[tid];
    y[tid + 256] = d1 * rstd * gamma[tid + 256] + beta[tid + 256];
    y[tid + 512] = d2 * rstd * gamma[tid + 512] + beta[tid + 512];
}

// ---------------------------------------------------------------------------
extern "C" void kernel_launch(void* const* d_in, const int* in_sizes, int n_in,
                              void* d_out, int out_size, void* d_ws, size_t ws_size,
                              hipStream_t stream) {
    const float* X    = (const float*)d_in[0];
    const float* mask = (const float*)d_in[1];
    const float* Wq   = (const float*)d_in[2];
    const float* bq   = (const float*)d_in[3];
    const float* Wk   = (const float*)d_in[4];
    const float* bk   = (const float*)d_in[5];
    const float* Wv   = (const float*)d_in[6];
    const float* bv   = (const float*)d_in[7];
    const float* Wo   = (const float*)d_in[8];
    const float* bo   = (const float*)d_in[9];
    const float* gamma = (const float*)d_in[10];
    const float* beta  = (const float*)d_in[11];

    const size_t QKV = (size_t)BATCH * NH * SEQ * DH;   // 6291456 halfs each
    _Float16* qws = (_Float16*)d_ws;
    _Float16* kws = qws + QKV;
    _Float16* vws = kws + QKV;
    _Float16* cws = vws + QKV;                          // [B*S, 768] f16
    float* Y = (float*)d_out;                           // pre-LN activation, LN in-place

    rba_qkv_kernel<<<dim3(NH, NROWS / 128, 3), 256, 0, stream>>>(
        X, Wq, bq, Wk, bk, Wv, bv, qws, kws, vws);
    rba_attn_kernel<<<dim3(SEQ / 64, BATCH * NH), 128, 0, stream>>>(
        qws, kws, vws, mask, cws);
    rba_proj_kernel<<<dim3(DMODEL / 64, NROWS / 128), 256, 0, stream>>>(
        cws, Wo, bo, X, Y);
    rba_ln_kernel<<<dim3(NROWS), 256, 0, stream>>>(Y, gamma, beta);
}
